// MultiHead_Attention_rel_36893769073026
// MI455X (gfx1250) — matrix operand order repaired and checked by reading, not compiled, not run on hardware
//
#include <hip/hip_runtime.h>
#include <hip/hip_bf16.h>

#define Bc  8
#define Lc  384
#define Hc  512
#define NHc 8
#define DHc 64

typedef __attribute__((ext_vector_type(16))) _Float16 v16h;
typedef __attribute__((ext_vector_type(8)))  _Float16 v8h;
typedef __attribute__((ext_vector_type(8)))  float    v8f;

static __device__ __forceinline__ v16h cvt16(float4 a, float4 b, float4 c, float4 d) {
  v16h f;
  f[0]=(_Float16)a.x;  f[1]=(_Float16)a.y;  f[2]=(_Float16)a.z;  f[3]=(_Float16)a.w;
  f[4]=(_Float16)b.x;  f[5]=(_Float16)b.y;  f[6]=(_Float16)b.z;  f[7]=(_Float16)b.w;
  f[8]=(_Float16)c.x;  f[9]=(_Float16)c.y;  f[10]=(_Float16)c.z; f[11]=(_Float16)c.w;
  f[12]=(_Float16)d.x; f[13]=(_Float16)d.y; f[14]=(_Float16)d.z; f[15]=(_Float16)d.w;
  return f;
}

// Generic GEMM: out = A[M,K] @ W[N,K]^T + bias[N] (+extra[N]); one wave per 16x16 tile.
// layout 0: row-major [M,N]; layout 1: head-major [B,NH,L,DH]; layout 2: [B,NH,DH,L].
__global__ void __launch_bounds__(128) gemm_wmma_kernel(
    const float* __restrict__ A, const float* __restrict__ W, const float* __restrict__ bias,
    void* __restrict__ out0, const float* __restrict__ extra0,
    void* __restrict__ out1, const float* __restrict__ extra1,
    int M, int N, int K, int layout, int outF32, int Lspan)
{
  int wid = blockIdx.x * (blockDim.x >> 5) + (threadIdx.x >> 5);
  int mt = (M + 15) >> 4;
  int nt = N >> 4;
  if (wid >= mt * nt) return;
  int tm = wid / nt, tn = wid % nt;
  int lane   = threadIdx.x & 31;
  int lrow   = lane & 15;
  int kbase  = (lane < 16) ? 0 : 8;   // unused; A-frag loads use kbB: {k..k+15}
  int kbB    = (lane < 16) ? 0 : 16;  // B-frag K base pattern {k..k+15}
  int rowOff = (lane < 16) ? 0 : 8;   // C/D layout row offset per VGPR

  int am = tm * 16 + lrow;
  bool av = am < M;
  const float* Arow = A + (size_t)(av ? am : 0) * K;
  const float* Wrow = W + (size_t)(tn * 16 + lrow) * K;

  v8f acc = {};
  for (int kk = 0; kk < K; kk += 32) {
    v16h af = {};
    if (av) {
      const float4* p = (const float4*)(Arow + kk + kbB);
      af = cvt16(p[0], p[1], p[2], p[3]);
    }
    const float4* q = (const float4*)(Wrow + kk + kbB);
    v16h bf = cvt16(q[0], q[1], q[2], q[3]);
    acc = __builtin_amdgcn_wmma_f32_16x16x32_f16(false, af, false, bf, (short)0, acc, false, false);
  }

  int col = tn * 16 + lrow;
  float b0 = bias[col] + (extra0 ? extra0[col] : 0.f);
  float b1 = extra1 ? (bias[col] + extra1[col]) : 0.f;
  #pragma unroll
  for (int r = 0; r < 8; ++r) {
    int row = tm * 16 + rowOff + r;
    if (row >= M) continue;
    size_t idx;
    if (layout == 0) {
      idx = (size_t)row * N + col;
    } else {
      int bb = row / Lspan, ii = row % Lspan;
      int hh = col >> 6, dd = col & 63;
      if (layout == 1) idx = (((size_t)(bb * NHc + hh)) * Lspan + ii) * (size_t)DHc + dd;
      else             idx = (((size_t)(bb * NHc + hh)) * DHc + dd) * (size_t)Lspan + ii;
    }
    float val = acc[r] + b0;
    if (outF32) {
      ((float*)out0)[idx] = val;
    } else {
      ((_Float16*)out0)[idx] = (_Float16)val;
      if (out1) ((_Float16*)out1)[idx] = (_Float16)(acc[r] + b1);
    }
  }
}

// Flash-style Transformer-XL attention: one wave per (b, h, 16-row i-tile).
__global__ void __launch_bounds__(32) attn_kernel(
    const _Float16* __restrict__ qu, const _Float16* __restrict__ qv,
    const _Float16* __restrict__ k16, const _Float16* __restrict__ vT,
    const _Float16* __restrict__ R, const int* __restrict__ seq_len,
    float* __restrict__ O)
{
  const int lane   = threadIdx.x & 31;
  const int lrow   = lane & 15;
  const int kbase  = (lane < 16) ? 0 : 8;
  const int kbB    = (lane < 16) ? 0 : 16;
  const int rowOff = (lane < 16) ? 0 : 8;

  int gid = blockIdx.x;
  int it = gid % (Lc / 16);
  int bh = gid / (Lc / 16);
  int h = bh % NHc, b = bh / NHc;
  int i0 = it * 16;
  int slen = seq_len[b];

  size_t headQ = ((size_t)(b * NHc + h)) * Lc * DHc;
  const _Float16* quB = qu + headQ;
  const _Float16* qvB = qv + headQ;
  const _Float16* kB  = k16 + headQ;
  const _Float16* vB  = vT + ((size_t)(b * NHc + h)) * DHc * Lc;

  __shared__ float sS[16][32];
  __shared__ float sT[16][48];
  __shared__ float sStat[16];

  // Preload (q+u) and (q+v) A-fragments for the whole i-tile (K = DH = 64 -> 2 frags each)
  v16h qaf[2], qvf[2];
  #pragma unroll
  for (int t = 0; t < 2; ++t) {
    const _Float16* p = quB + ((size_t)(i0 + lrow)) * DHc + t * 32 + kbB;
    v8h lo = *(const v8h*)p;
    v8h hi = *(const v8h*)(p + 8);
    v16h f;
    #pragma unroll
    for (int e = 0; e < 8; ++e) { f[e] = lo[e]; f[8 + e] = hi[e]; }
    qaf[t] = f;
    const _Float16* p2 = qvB + ((size_t)(i0 + lrow)) * DHc + t * 32 + kbB;
    v8h lo2 = *(const v8h*)p2;
    v8h hi2 = *(const v8h*)(p2 + 8);
    v16h g;
    #pragma unroll
    for (int e = 0; e < 8; ++e) { g[e] = lo2[e]; g[8 + e] = hi2[e]; }
    qvf[t] = g;
  }

  float m_run = -1e30f, l_run = 0.f;
  v8f o0 = {}, o1 = {}, o2 = {}, o3 = {};

  for (int j0 = 0; j0 < Lc; j0 += 32) {
    int rbase = j0 - i0 + (Lc - 15);   // min 1, max+47 = 768: always inside [0, 2L]
    v8f c0 = {}, c1 = {};
    v8f t0 = {}, t1 = {}, t2 = {};
    #pragma unroll
    for (int t = 0; t < 2; ++t) {
      // content term: (q+u) . k^T  (B-frag: lane = column j, contiguous dh)
      v16h bk0 = *(const v16h*)(kB + ((size_t)(j0 + lrow))      * DHc + t * 32 + kbB);
      v16h bk1 = *(const v16h*)(kB + ((size_t)(j0 + 16 + lrow)) * DHc + t * 32 + kbB);
      c0 = __builtin_amdgcn_wmma_f32_16x16x32_f16(false, qaf[t], false, bk0, (short)0, c0, false, false);
      c1 = __builtin_amdgcn_wmma_f32_16x16x32_f16(false, qaf[t], false, bk1, (short)0, c1, false, false);
      // position term window: (q+v) . R^T over 48 relative positions
      const _Float16* rb = R + (size_t)h * DHc + t * 32 + kbB;
      v16h br0 = *(const v16h*)(rb + ((size_t)(rbase + lrow))      * Hc);
      v16h br1 = *(const v16h*)(rb + ((size_t)(rbase + 16 + lrow)) * Hc);
      v16h br2 = *(const v16h*)(rb + ((size_t)(rbase + 32 + lrow)) * Hc);
      t0 = __builtin_amdgcn_wmma_f32_16x16x32_f16(false, qvf[t], false, br0, (short)0, t0, false, false);
      t1 = __builtin_amdgcn_wmma_f32_16x16x32_f16(false, qvf[t], false, br1, (short)0, t1, false, false);
      t2 = __builtin_amdgcn_wmma_f32_16x16x32_f16(false, qvf[t], false, br2, (short)0, t2, false, false);
    }
    // Stage C-layout tiles to LDS so each lane can gather its softmax row + diagonal shift
    #pragma unroll
    for (int r = 0; r < 8; ++r) {
      sS[rowOff + r][lrow]      = c0[r];
      sS[rowOff + r][16 + lrow] = c1[r];
      sT[rowOff + r][lrow]      = t0[r];
      sT[rowOff + r][16 + lrow] = t1[r];
      sT[rowOff + r][32 + lrow] = t2[r];
    }
    __syncthreads();

    // Lane owns row i = lrow with the 16 j-columns of its future A-fragment slots.
    float vals[16];
    float tmax = -1e30f;
    #pragma unroll
    for (int e = 0; e < 16; ++e) {
      int jj = kbB + e;             // {kbB..kbB+15}
      float s = (sS[lrow][jj] + sT[lrow][jj - lrow + 15]) * 0.125f;  // 1/sqrt(64)
      if (j0 + jj >= slen) s = -1e15f;
      vals[e] = s;
      tmax = fmaxf(tmax, s);
    }
    tmax = fmaxf(tmax, __shfl_xor(tmax, 16, 32));
    float mn = fmaxf(m_run, tmax);
    float corr = __expf(m_run - mn);
    float ps = 0.f;
    v16h pf;
    #pragma unroll
    for (int e = 0; e < 16; ++e) {
      float p = __expf(vals[e] - mn);
      ps += p;
      pf[e] = (_Float16)p;
    }
    ps += __shfl_xor(ps, 16, 32);
    l_run = l_run * corr + ps;
    m_run = mn;

    __syncthreads();
    if (lane < 16) sStat[lrow] = corr;
    __syncthreads();
    float fr[8];
    #pragma unroll
    for (int r = 0; r < 8; ++r) fr[r] = sStat[rowOff + r];
    #pragma unroll
    for (int r = 0; r < 8; ++r) { o0[r] *= fr[r]; o1[r] *= fr[r]; o2[r] *= fr[r]; o3[r] *= fr[r]; }

    // P.V : vT layout makes B-fragments contiguous 32B loads per lane
    v16h bv0 = *(const v16h*)(vB + ((size_t)( 0 + lrow)) * Lc + j0 + kbB);
    v16h bv1 = *(const v16h*)(vB + ((size_t)(16 + lrow)) * Lc + j0 + kbB);
    v16h bv2 = *(const v16h*)(vB + ((size_t)(32 + lrow)) * Lc + j0 + kbB);
    v16h bv3 = *(const v16h*)(vB + ((size_t)(48 + lrow)) * Lc + j0 + kbB);
    o0 = __builtin_amdgcn_wmma_f32_16x16x32_f16(false, pf, false, bv0, (short)0, o0, false, false);
    o1 = __builtin_amdgcn_wmma_f32_16x16x32_f16(false, pf, false, bv1, (short)0, o1, false, false);
    o2 = __builtin_amdgcn_wmma_f32_16x16x32_f16(false, pf, false, bv2, (short)0, o2, false, false);
    o3 = __builtin_amdgcn_wmma_f32_16x16x32_f16(false, pf, false, bv3, (short)0, o3, false, false);
    __syncthreads();
  }

  __syncthreads();
  if (lane < 16) sStat[lrow] = 1.f / l_run;
  __syncthreads();
  #pragma unroll
  for (int r = 0; r < 8; ++r) {
    float inv = sStat[rowOff + r];
    int row = i0 + rowOff + r;
    size_t base = ((size_t)b * Lc + row) * Hc + (size_t)h * DHc + lrow;
    O[base + 0]  = o0[r] * inv;
    O[base + 16] = o1[r] * inv;
    O[base + 32] = o2[r] * inv;
    O[base + 48] = o3[r] * inv;
  }
}

extern "C" void kernel_launch(void* const* d_in, const int* in_sizes, int n_in,
                              void* d_out, int out_size, void* d_ws, size_t ws_size,
                              hipStream_t stream) {
  (void)in_sizes; (void)n_in; (void)out_size; (void)ws_size;
  const float* key     = (const float*)d_in[0];
  const float* query   = (const float*)d_in[1];
  const float* value   = (const float*)d_in[2];
  const int*   seq_len = (const int*)  d_in[3];
  const float* pe      = (const float*)d_in[4];
  const float* Wk      = (const float*)d_in[5];
  const float* bk      = (const float*)d_in[6];
  const float* Wq      = (const float*)d_in[7];
  const float* bq      = (const float*)d_in[8];
  const float* Wv      = (const float*)d_in[9];
  const float* bv      = (const float*)d_in[10];
  const float* Wr      = (const float*)d_in[11];
  const float* br      = (const float*)d_in[12];
  const float* u_bias  = (const float*)d_in[13];
  const float* v_bias  = (const float*)d_in[14];
  const float* Wff     = (const float*)d_in[15];
  const float* bff     = (const float*)d_in[16];

  char* ws = (char*)d_ws;
  size_t off = 0;
  auto alloc = [&](size_t bytes) -> void* {
    void* p = (void*)(ws + off);
    off += (bytes + 255) & ~(size_t)255;
    return p;
  };
  const size_t QSZ = (size_t)Bc * NHc * Lc * DHc * sizeof(_Float16);
  _Float16* qu16 = (_Float16*)alloc(QSZ);
  _Float16* qv16 = (_Float16*)alloc(QSZ);
  _Float16* k16  = (_Float16*)alloc(QSZ);
  _Float16* vT16 = (_Float16*)alloc(QSZ);
  _Float16* R16  = (_Float16*)alloc((size_t)(2 * Lc + 1) * Hc * sizeof(_Float16));
  float*    O32  = (float*)   alloc((size_t)Bc * Lc * Hc * sizeof(float));

  const int M = Bc * Lc;                       // 3072
  const int tiles = (M / 16) * (Hc / 16);      // 6144
  dim3 blk(128);
  int grid = (tiles + 3) / 4;

  // q -> qu = q+u, qv = q+v (head-major f16)
  gemm_wmma_kernel<<<grid, blk, 0, stream>>>(query, Wq, bq, qu16, u_bias, qv16, v_bias,
                                             M, Hc, Hc, 1, 0, Lc);
  // k (head-major f16)
  gemm_wmma_kernel<<<grid, blk, 0, stream>>>(key, Wk, bk, k16, nullptr, nullptr, nullptr,
                                             M, Hc, Hc, 1, 0, Lc);
  // v (head-major transposed f16)
  gemm_wmma_kernel<<<grid, blk, 0, stream>>>(value, Wv, bv, vT16, nullptr, nullptr, nullptr,
                                             M, Hc, Hc, 2, 0, Lc);
  // R = pe @ Wr^T + br  (f16 row-major, 769 rows -> padded tiles)
  int Mr = 2 * Lc + 1;
  int tilesR = ((Mr + 15) / 16) * (Hc / 16);
  gemm_wmma_kernel<<<(tilesR + 3) / 4, blk, 0, stream>>>(pe, Wr, br, R16, nullptr, nullptr, nullptr,
                                                         Mr, Hc, Hc, 0, 0, 0);
  // attention
  attn_kernel<<<Bc * NHc * (Lc / 16), dim3(32), 0, stream>>>(qu16, qv16, k16, vT16, R16,
                                                             seq_len, O32);
  // final FF: d_out = O @ Wff^T + bff (f32 out)
  gemm_wmma_kernel<<<grid, blk, 0, stream>>>(O32, Wff, bff, d_out, nullptr, nullptr, nullptr,
                                             M, Hc, Hc, 0, 1, 0);
}
